// SNN_7447473291682
// MI455X (gfx1250) — matrix operand order repaired and checked by reading, not compiled, not run on hardware
//
#include <hip/hip_runtime.h>

// SNN (norse LIF recurrent + LI readout) forward on MI455X (gfx1250, wave32).
// xi = x @ w_in^T        : 12800 x 2048 x 8192 bf16 WMMA GEMM (f32 accum),
//                          double-buffered LDS + async global->LDS weight tiles.
// per step: z @ w_rec^T  : 64 x 2048 x 2048 bf16 WMMA GEMM + fused LIF update,
//                          both tiles via async global->LDS.
// readout                : wave-per-(b,o) dot + LI cell, writes vo to d_out.

#define T_STEPS 200
#define BATCH   64
#define IN_FEAT 8192
#define HID     2048
#define OUT_F   20

typedef __attribute__((ext_vector_type(16))) __bf16         v16bf;
typedef __attribute__((ext_vector_type(8)))  float          v8f;
typedef __attribute__((ext_vector_type(8)))  unsigned short u16x8;
typedef __attribute__((ext_vector_type(4)))  float          f32x4;
typedef __attribute__((ext_vector_type(4)))  int            i32x4;

union FragAB { v16bf v; u16x8 h[2]; };

// ---- CDNA5 async global->LDS path (guarded; falls back to sync staging) ----
#if defined(__has_builtin)
#if __has_builtin(__builtin_amdgcn_global_load_async_to_lds_b128) && \
    __has_builtin(__builtin_amdgcn_s_wait_asynccnt)
#define USE_ASYNC 1
#endif
#endif
#ifndef USE_ASYNC
#define USE_ASYNC 0
#endif

#if USE_ASYNC
// probe-derived signature: (int4 addrspace(1)*, int4 addrspace(3)*, imm offset, imm cpol)
typedef __attribute__((address_space(1))) i32x4* glb_b128p;
typedef __attribute__((address_space(3))) i32x4* lds_b128p;
#define ASYNC_B128(g, l) \
  __builtin_amdgcn_global_load_async_to_lds_b128((glb_b128p)(g), (lds_b128p)(l), 0, 0)
#define WAIT_ASYNC() __builtin_amdgcn_s_wait_asynccnt(0)
#else
#define WAIT_ASYNC()
#endif

__device__ __forceinline__ void cvt_store16(const f32x4* a, __bf16* dst) {
  union { __bf16 b[16]; u16x8 v[2]; } cv;
#pragma unroll
  for (int q = 0; q < 4; ++q) {
    cv.b[4*q+0] = (__bf16)a[q].x; cv.b[4*q+1] = (__bf16)a[q].y;
    cv.b[4*q+2] = (__bf16)a[q].z; cv.b[4*q+3] = (__bf16)a[q].w;
  }
  *(u16x8*)dst       = cv.v[0];
  *(u16x8*)(dst + 8) = cv.v[1];
}

// ---------------- utility kernels ----------------

__global__ __launch_bounds__(256) void k_zero4(f32x4* __restrict__ p, int n4) {
  int i = blockIdx.x * 256 + threadIdx.x;
  if (i < n4) {
    f32x4 z = {0.f, 0.f, 0.f, 0.f};
    p[i] = z;
  }
}

__global__ __launch_bounds__(256) void k_f32_to_bf16(const float* __restrict__ src,
                                                     __bf16* __restrict__ dst, int n8) {
  int i = blockIdx.x * 256 + threadIdx.x;
  if (i >= n8) return;
  const f32x4* s = (const f32x4*)src + (size_t)i * 2;
  f32x4 a[2]; a[0] = s[0]; a[1] = s[1];
  union { __bf16 b[8]; u16x8 v; } cv;
  cv.b[0] = (__bf16)a[0].x; cv.b[1] = (__bf16)a[0].y; cv.b[2] = (__bf16)a[0].z; cv.b[3] = (__bf16)a[0].w;
  cv.b[4] = (__bf16)a[1].x; cv.b[5] = (__bf16)a[1].y; cv.b[6] = (__bf16)a[1].z; cv.b[7] = (__bf16)a[1].w;
  *(u16x8*)(dst + (size_t)i * 8) = cv.v;
}

// ---------------- input projection GEMM ----------------
// Y[m][n] = sum_k X[m][k] * W[n][k];  X fp32 (cvt to bf16 in staging), W bf16.
// Block tile 128(M) x 128(N) x 32(K); 8 waves 4(M)x2(N); wave -> 2x4 16x16 tiles.
// Double-buffered LDS; W tiles async global->LDS; grid.x = N-blocks for A reuse.

__global__ __launch_bounds__(256) void k_gemm_in(const float* __restrict__ X,
                                                 const __bf16* __restrict__ W,
                                                 float* __restrict__ Y,
                                                 int N, int K) {
  __shared__ __bf16 As[2][128][32];
  __shared__ __bf16 Bs[2][128][32];

  const int tid  = threadIdx.x;
  const int lane = tid & 31;
  const int wave = tid >> 5;
  const int n0 = blockIdx.x * 128;  // N fastest-varying: 16 blocks share one A slab
  const int m0 = blockIdx.y * 128;
  const int wm = (wave & 3) * 32;
  const int wn = (wave >> 2) * 64;

  const v8f zero8 = {0.f, 0.f, 0.f, 0.f, 0.f, 0.f, 0.f, 0.f};
  v8f acc[2][4];
#pragma unroll
  for (int i = 0; i < 2; ++i)
#pragma unroll
    for (int j = 0; j < 4; ++j) acc[i][j] = zero8;

  const int ra = tid >> 1;         // staging row (128 rows, 2 threads/row)
  const int ca = (tid & 1) * 16;   // 16 elems each
  const int lr = lane & 15, hi = lane >> 4;
  const int aOff = hi * 8;         // A frag: lanes>=16 start at K=8
  const int bOff = hi * 8;         // B frag: lanes>=16 start at K=8

  const float*  gA = X + (size_t)(m0 + ra) * K + ca;
  const __bf16* gB = W + (size_t)(n0 + ra) * K + ca;

  f32x4 a[4];
  // prologue: stage K-tile 0 into buffer 0
  {
    const f32x4* pa = (const f32x4*)gA;
    a[0] = pa[0]; a[1] = pa[1]; a[2] = pa[2]; a[3] = pa[3];
#if USE_ASYNC
    ASYNC_B128(gB,     &Bs[0][ra][ca]);
    ASYNC_B128(gB + 8, &Bs[0][ra][ca + 8]);
#else
    const u16x8* pb = (const u16x8*)gB;
    *(u16x8*)&Bs[0][ra][ca]     = pb[0];
    *(u16x8*)&Bs[0][ra][ca + 8] = pb[1];
#endif
    cvt_store16(a, &As[0][ra][ca]);
  }
  WAIT_ASYNC();
  __syncthreads();

  for (int k0 = 0; k0 < K; k0 += 32) {
    const int cur = (k0 >> 5) & 1;
    const int nxt = cur ^ 1;
    const bool more = (k0 + 32) < K;
    if (more) {  // prefetch next K-tile while computing current
      const f32x4* pa = (const f32x4*)(gA + k0 + 32);
      a[0] = pa[0]; a[1] = pa[1]; a[2] = pa[2]; a[3] = pa[3];
#if USE_ASYNC
      ASYNC_B128(gB + k0 + 32, &Bs[nxt][ra][ca]);
      ASYNC_B128(gB + k0 + 40, &Bs[nxt][ra][ca + 8]);
#else
      const u16x8* pb = (const u16x8*)(gB + k0 + 32);
      *(u16x8*)&Bs[nxt][ra][ca]     = pb[0];
      *(u16x8*)&Bs[nxt][ra][ca + 8] = pb[1];
#endif
    }

    FragAB fa[2];
#pragma unroll
    for (int mi = 0; mi < 2; ++mi) {
      const __bf16* p = &As[cur][wm + mi * 16 + lr][aOff];
      fa[mi].h[0] = *(const u16x8*)p;          // K = aOff .. aOff+7
      fa[mi].h[1] = *(const u16x8*)(p + 16);   // K = 16+aOff ..
    }
#pragma unroll
    for (int ni = 0; ni < 4; ++ni) {
      FragAB fb;
      const __bf16* q = &Bs[cur][wn + ni * 16 + lr][bOff];
      fb.h[0] = *(const u16x8*)q;
      fb.h[1] = *(const u16x8*)(q + 16);
#pragma unroll
      for (int mi = 0; mi < 2; ++mi)
        acc[mi][ni] = __builtin_amdgcn_wmma_f32_16x16x32_bf16(
            false, fa[mi].v, false, fb.v, (short)0, acc[mi][ni], false, false);
    }

    if (more) cvt_store16(a, &As[nxt][ra][ca]);
    WAIT_ASYNC();
    __syncthreads();
  }

  const int rb = hi * 8;  // D layout: lanes>=16 hold M rows 8..15
#pragma unroll
  for (int mi = 0; mi < 2; ++mi)
#pragma unroll
    for (int ni = 0; ni < 4; ++ni) {
      float* py = Y + (size_t)(m0 + wm + mi * 16 + rb) * N + (n0 + wn + ni * 16 + lr);
#pragma unroll
      for (int r = 0; r < 8; ++r) py[(size_t)r * N] = acc[mi][ni][r];
    }
}

// ---------------- recurrent step: GEMM + fused LIF ----------------
// rec[b][h] = sum_j Z[b][j] * Wr[h][j]; LIF update fused in epilogue.
// Block tile 64(M=batch) x 64(N) x 32(K); 8 waves 4(M)x2(N); wave -> 1x2 tiles.
// Both tiles bf16 -> async global->LDS, double-buffered.

__global__ __launch_bounds__(256) void k_step(const __bf16* __restrict__ Z,
                                              const __bf16* __restrict__ Wr,
                                              const float* __restrict__ XIt,
                                              float* __restrict__ V,
                                              float* __restrict__ I,
                                              __bf16* __restrict__ Znext) {
  __shared__ __bf16 As[2][64][32];
  __shared__ __bf16 Bs[2][64][32];

  const int tid  = threadIdx.x;
  const int lane = tid & 31;
  const int wave = tid >> 5;
  const int n0 = blockIdx.x * 64;
  const int wm = (wave & 3) * 16;
  const int wn = (wave >> 2) * 32;
  const int lr = lane & 15, hi = lane >> 4;
  const int aOff = hi * 8, bOff = hi * 8;

  const v8f zero8 = {0.f, 0.f, 0.f, 0.f, 0.f, 0.f, 0.f, 0.f};
  v8f acc[2];
  acc[0] = zero8; acc[1] = zero8;

  const int rs = tid >> 2;          // 64 rows, 4 threads/row
  const int cs = (tid & 3) * 8;     // 8 elems (16B) each
  const __bf16* gz = Z  + (size_t)rs * HID + cs;
  const __bf16* gw = Wr + (size_t)(n0 + rs) * HID + cs;

  // prologue
#if USE_ASYNC
  ASYNC_B128(gz, &As[0][rs][cs]);
  ASYNC_B128(gw, &Bs[0][rs][cs]);
#else
  *(u16x8*)&As[0][rs][cs] = *(const u16x8*)gz;
  *(u16x8*)&Bs[0][rs][cs] = *(const u16x8*)gw;
#endif
  WAIT_ASYNC();
  __syncthreads();

  for (int k0 = 0; k0 < HID; k0 += 32) {
    const int cur = (k0 >> 5) & 1;
    const int nxt = cur ^ 1;
    const bool more = (k0 + 32) < HID;
    if (more) {
#if USE_ASYNC
      ASYNC_B128(gz + k0 + 32, &As[nxt][rs][cs]);
      ASYNC_B128(gw + k0 + 32, &Bs[nxt][rs][cs]);
#else
      *(u16x8*)&As[nxt][rs][cs] = *(const u16x8*)(gz + k0 + 32);
      *(u16x8*)&Bs[nxt][rs][cs] = *(const u16x8*)(gw + k0 + 32);
#endif
    }

    FragAB fa;
    const __bf16* p = &As[cur][wm + lr][aOff];
    fa.h[0] = *(const u16x8*)p;
    fa.h[1] = *(const u16x8*)(p + 16);
#pragma unroll
    for (int ni = 0; ni < 2; ++ni) {
      FragAB fb;
      const __bf16* q = &Bs[cur][wn + ni * 16 + lr][bOff];
      fb.h[0] = *(const u16x8*)q;
      fb.h[1] = *(const u16x8*)(q + 16);
      acc[ni] = __builtin_amdgcn_wmma_f32_16x16x32_bf16(
          false, fa.v, false, fb.v, (short)0, acc[ni], false, false);
    }
    WAIT_ASYNC();
    __syncthreads();
  }

  // fused LIF update (norse lif_step): each (b,h) of this N-slice touched once
  const int rb = hi * 8;
#pragma unroll
  for (int ni = 0; ni < 2; ++ni) {
#pragma unroll
    for (int r = 0; r < 8; ++r) {
      const int b = wm + rb + r;
      const int h = n0 + wn + ni * 16 + lr;
      const size_t idx = (size_t)b * HID + h;
      float v   = V[idx];
      float cur = I[idx];
      float v_dec = v + 0.1f * (cur - v);        // DT*TAU_MEM_INV = 0.1
      float i_dec = cur * 0.8f;                  // 1 - DT*TAU_SYN_INV
      float z = (v_dec > 0.3f) ? 1.0f : 0.0f;    // heaviside(v_dec - V_TH)
      V[idx] = (1.0f - z) * v_dec;
      I[idx] = i_dec + XIt[idx] + acc[ni][r];
      Znext[idx] = (__bf16)z;                    // exact: z in {0,1}
    }
  }
}

// ---------------- readout: z @ w_out^T + LI cell ----------------

__global__ __launch_bounds__(256) void k_readout(const __bf16* __restrict__ Z,
                                                 const float* __restrict__ Wo,
                                                 float* __restrict__ VO,
                                                 float* __restrict__ IO,
                                                 float* __restrict__ Ot) {
  const int task = blockIdx.x * 8 + (threadIdx.x >> 5);  // 0 .. B*OUT-1
  const int lane = threadIdx.x & 31;
  const int b = task / OUT_F;
  const int o = task - b * OUT_F;
  const __bf16* z = Z  + (size_t)b * HID;
  const float*  w = Wo + (size_t)o * HID;
  float s = 0.f;
#pragma unroll 4
  for (int k = lane; k < HID; k += 32) s += (float)z[k] * w[k];
#pragma unroll
  for (int off = 16; off > 0; off >>= 1) s += __shfl_xor(s, off, 32);
  if (lane == 0) {
    float vo = VO[task], io = IO[task];
    float vo_new = vo + 0.1f * (io - vo);  // uses io BEFORE adding inp
    IO[task] = io * 0.8f + s;
    VO[task] = vo_new;
    Ot[task] = vo_new;
  }
}

// ---------------- host launcher ----------------

extern "C" void kernel_launch(void* const* d_in, const int* in_sizes, int n_in,
                              void* d_out, int out_size, void* d_ws, size_t ws_size,
                              hipStream_t stream) {
  (void)in_sizes; (void)n_in; (void)out_size; (void)ws_size;
  const float* x     = (const float*)d_in[0];  // [T,B,C,H,W] -> [T*B, 8192]
  const float* w_in  = (const float*)d_in[1];  // [2048, 8192]
  const float* w_rec = (const float*)d_in[2];  // [2048, 2048]
  const float* w_out = (const float*)d_in[3];  // [20, 2048]
  float* out = (float*)d_out;                  // [T, B, 20]

  char* p = (char*)d_ws;
  float*  xi      = (float*)p;  p += (size_t)T_STEPS * BATCH * HID * sizeof(float);   // 104.8 MB
  __bf16* w_in_b  = (__bf16*)p; p += (size_t)HID * IN_FEAT * sizeof(__bf16);          // 33.5 MB
  __bf16* w_rec_b = (__bf16*)p; p += (size_t)HID * HID * sizeof(__bf16);              //  8.4 MB
  __bf16* zbuf    = (__bf16*)p; p += (size_t)2 * BATCH * HID * sizeof(__bf16);        //  0.5 MB
  float*  V       = (float*)p;  p += (size_t)BATCH * HID * sizeof(float);
  float*  I       = (float*)p;  p += (size_t)BATCH * HID * sizeof(float);
  float*  VO      = (float*)p;  p += (size_t)BATCH * OUT_F * sizeof(float);
  float*  IO      = (float*)p;  p += (size_t)BATCH * OUT_F * sizeof(float);

  // zero all state (zbuf, V, I, VO, IO are contiguous)
  {
    size_t bytes = (size_t)((char*)p - (char*)zbuf);
    int n4 = (int)(bytes / 16);
    k_zero4<<<(n4 + 255) / 256, 256, 0, stream>>>((f32x4*)zbuf, n4);
  }
  // weight conversions
  {
    int n8 = HID * IN_FEAT / 8;
    k_f32_to_bf16<<<(n8 + 255) / 256, 256, 0, stream>>>(w_in, w_in_b, n8);
  }
  {
    int n8 = HID * HID / 8;
    k_f32_to_bf16<<<(n8 + 255) / 256, 256, 0, stream>>>(w_rec, w_rec_b, n8);
  }

  // input projection: xi = x @ w_in^T   (M=12800, N=2048, K=8192)
  // grid.x = N-blocks (fastest) so the 16 consumers of each A slab co-schedule.
  dim3 g1(HID / 128, T_STEPS * BATCH / 128);
  k_gemm_in<<<g1, 256, 0, stream>>>(x, w_in_b, xi, HID, IN_FEAT);

  // sequential scan
  for (int t = 0; t < T_STEPS; ++t) {
    const __bf16* zprev = zbuf + (size_t)(t & 1) * BATCH * HID;
    __bf16*       znext = zbuf + (size_t)((t + 1) & 1) * BATCH * HID;
    k_step<<<HID / 64, 256, 0, stream>>>(zprev, w_rec_b,
                                         xi + (size_t)t * BATCH * HID, V, I, znext);
    k_readout<<<BATCH * OUT_F / 8, 256, 0, stream>>>(znext, w_out, VO, IO,
                                                     out + (size_t)t * BATCH * OUT_F);
  }
}
